// ArgumentGCN_wreverseedges_double_7885559956056
// MI455X (gfx1250) — matrix operand order repaired and checked by reading, not compiled, not run on hardware
//
#include <hip/hip_runtime.h>
#include <stdint.h>

// Problem dims (fixed by the reference)
#define DDIM 768
#define NN   256
#define NB   64
#define ROWS (NB * NN)        // 16384 rows of "node"
#define NCAT (5 * DDIM)       // 3840: [self | a | p | ar | pr] projections

// GEMM tiling: 256 threads (8 waves), 128x128 C tile, wave tile 32x64, K-step 32
#define BM 128
#define BN 128
#define BK 32
#define LDT 40                // LDS pitch in halfs (80B rows: conflict-spread, 16B aligned)

typedef __attribute__((ext_vector_type(16))) __bf16 v16bf;
typedef __attribute__((ext_vector_type(8)))  float  v8f;
typedef int v4i __attribute__((vector_size(16)));

union FragU { v16bf v; uint4 q[2]; };

__device__ __forceinline__ uint16_t f2bf(float f) {
  uint32_t u = __float_as_uint(f);
  return (uint16_t)((u + 0x7FFFu + ((u >> 16) & 1u)) >> 16);  // RNE
}
__device__ __forceinline__ float bf2f(uint16_t h) {
  return __uint_as_float(((uint32_t)h) << 16);
}

// ---- gfx1250 async global->LDS copy (ASYNCcnt path), guarded fallback ----
#if defined(__has_builtin)
# if __has_builtin(__builtin_amdgcn_global_load_async_to_lds_b128) && \
     __has_builtin(__builtin_amdgcn_s_wait_asynccnt)
#  define HAVE_ASYNC_LDS 1
# else
#  define HAVE_ASYNC_LDS 0
# endif
#else
# define HAVE_ASYNC_LDS 0
#endif

#if HAVE_ASYNC_LDS
typedef __attribute__((address_space(1))) v4i* as1_v4i_ptr;
typedef __attribute__((address_space(3))) v4i* as3_v4i_ptr;
#endif

__device__ __forceinline__ void copy16_g2lds(const uint16_t* g, uint16_t* l) {
#if HAVE_ASYNC_LDS
  __builtin_amdgcn_global_load_async_to_lds_b128((as1_v4i_ptr)g, (as3_v4i_ptr)l, 0, 0);
#else
  *(uint4*)l = *(const uint4*)g;
#endif
}
__device__ __forceinline__ void async_wait() {
#if HAVE_ASYNC_LDS
  __builtin_amdgcn_s_wait_asynccnt(0);
#endif
}

// ---------------- prep: dw = sigmoid(node.Wq + bq); dwm = dw*mask ----------------
__global__ __launch_bounds__(256) void k_dw(const float* __restrict__ node,
                                            const int* __restrict__ nmask,
                                            const float* __restrict__ Wq,
                                            const float* __restrict__ bq,
                                            float* __restrict__ dwm,
                                            float* __restrict__ outW) {
  const int row  = blockIdx.x * 8 + (threadIdx.x >> 5);
  const int lane = threadIdx.x & 31;
  const float* p = node + (size_t)row * DDIM;
  float s = 0.f;
  for (int q = 0; q < DDIM; q += 32) s += p[q + lane] * Wq[q + lane];
  for (int o = 16; o > 0; o >>= 1) s += __shfl_xor(s, o, 32);
  if (lane == 0) {
    float dw = 1.f / (1.f + __expf(-(s + bq[0])));
    outW[row] = dw;                       // all_weight output [B,1,N] flat
    dwm[row]  = dw * (float)nmask[row];   // fold mask_j into dw
  }
}

// ---------------- prep: rden = 1/max(neigh,1) ----------------
__global__ __launch_bounds__(256) void k_denom(const int* __restrict__ ag,
                                               const int* __restrict__ pg,
                                               const int* __restrict__ nmask,
                                               float* __restrict__ rden) {
  const int b = blockIdx.x >> 8;
  const int i = blockIdx.x & 255;
  const int j = threadIdx.x;
  const int gb = b * NN * NN;
  float mi = (float)nmask[b * NN + i], mj = (float)nmask[b * NN + j];
  float e = (i == j) ? 0.f
            : mi * mj * (float)(ag[gb + i * NN + j] + pg[gb + i * NN + j] +
                                ag[gb + j * NN + i] + pg[gb + j * NN + i]);
  __shared__ float red[256];
  red[j] = e; __syncthreads();
  for (int s = 128; s > 0; s >>= 1) { if (j < s) red[j] += red[j + s]; __syncthreads(); }
  if (j == 0) { float n = red[0]; rden[blockIdx.x] = 1.f / (n >= 1.f ? n : 1.f); }
}

// ---------------- prep: Wcat(bf16) = [Ws; Wa+Wa2; Wp+Wp2; Wap+Wap2; Wpp+Wpp2] ----------------
__global__ __launch_bounds__(256) void k_wcat(const float* __restrict__ Ws,
    const float* __restrict__ Wa,  const float* __restrict__ Wp,
    const float* __restrict__ Wap, const float* __restrict__ Wpp,
    const float* __restrict__ Wa2, const float* __restrict__ Wp2,
    const float* __restrict__ Wap2,const float* __restrict__ Wpp2,
    uint16_t* __restrict__ Wcat) {
  int idx = blockIdx.x * 256 + threadIdx.x;      // 0 .. 3840*768-1
  int r = idx / DDIM, c = idx - r * DDIM;
  int g = r / DDIM, rr = r - g * DDIM;
  size_t o = (size_t)rr * DDIM + c;
  float v;
  if      (g == 0) v = Ws[o];
  else if (g == 1) v = Wa[o]  + Wa2[o];
  else if (g == 2) v = Wp[o]  + Wp2[o];
  else if (g == 3) v = Wap[o] + Wap2[o];
  else             v = Wpp[o] + Wpp2[o];
  Wcat[idx] = f2bf(v);
}

// ---------------- GEMM1: P = bf16(node) @ Wcat^T ----------------
// self slice -> PselfT[e][row] (transposed); relation slices -> PT[b][g][e][j] (transposed)
__global__ __launch_bounds__(256) void k_gemm1(const float* __restrict__ node,
                                               const uint16_t* __restrict__ Wcat,
                                               uint16_t* __restrict__ PselfT,
                                               uint16_t* __restrict__ PT) {
  __shared__ uint16_t At[2][BM * LDT];
  __shared__ uint16_t Bt[2][BN * LDT];
  const int tid = threadIdx.x;
  const int m0 = blockIdx.y * BM, n0 = blockIdx.x * BN;
  const int lane = tid & 31, wave = tid >> 5;
  const int wm = (wave & 3) * 32, wn = (wave >> 2) * 64;
  const int lr = lane & 15, hi = lane >> 4;
  const int a_k2 = (tid & 15) * 2, a_r = tid >> 4;   // A staging: 2 halfs/thread
  const int b_k8 = (tid & 3) * 8,  b_r = tid >> 2;   // B staging: 8 halfs/thread

  v8f zero = {0.f,0.f,0.f,0.f,0.f,0.f,0.f,0.f};
  v8f acc[2][4];
  #pragma unroll
  for (int mi = 0; mi < 2; ++mi)
    #pragma unroll
    for (int ni = 0; ni < 4; ++ni) acc[mi][ni] = zero;

  auto stage = [&](int kb, int buf) {
    float2 fa[8];                                    // all global loads in flight first
    #pragma unroll
    for (int p = 0; p < 8; ++p)
      fa[p] = *(const float2*)&node[(size_t)(m0 + a_r + p * 16) * DDIM + kb + a_k2];
    #pragma unroll
    for (int p = 0; p < 2; ++p) {                    // B: async 16B copies
      int nl = b_r + p * 64;
      copy16_g2lds(&Wcat[(size_t)(n0 + nl) * DDIM + kb + b_k8], &Bt[buf][nl * LDT + b_k8]);
    }
    #pragma unroll
    for (int p = 0; p < 8; ++p) {                    // convert + packed b32 LDS stores
      uint32_t pk = (uint32_t)f2bf(fa[p].x) | ((uint32_t)f2bf(fa[p].y) << 16);
      *(uint32_t*)&At[buf][(a_r + p * 16) * LDT + a_k2] = pk;
    }
  };

  stage(0, 0);
  const int nT = DDIM / BK;
  for (int it = 0; it < nT; ++it) {
    async_wait();
    __syncthreads();
    const int cur = it & 1;
    if (it + 1 < nT) stage((it + 1) * BK, cur ^ 1);  // overlap next tile with WMMAs
    v16bf af[2], bfv[4];
    #pragma unroll
    for (int mi = 0; mi < 2; ++mi) {                 // ISA 16-bit A layout
      const uint16_t* rp = &At[cur][(wm + mi * 16 + lr) * LDT];
      FragU u;
      u.q[0] = *(const uint4*)(rp + (hi ? 8 : 0));
      u.q[1] = *(const uint4*)(rp + (hi ? 24 : 16));
      af[mi] = u.v;
    }
    #pragma unroll
    for (int ni = 0; ni < 4; ++ni) {                 // B: ISA 16-bit layout (same k order as A)
      const uint16_t* cp = &Bt[cur][(wn + ni * 16 + lr) * LDT + hi * 8];
      FragU u;
      u.q[0] = *(const uint4*)(cp);
      u.q[1] = *(const uint4*)(cp + 16);
      bfv[ni] = u.v;
    }
    #pragma unroll
    for (int mi = 0; mi < 2; ++mi)
      #pragma unroll
      for (int ni = 0; ni < 4; ++ni)
        acc[mi][ni] = __builtin_amdgcn_wmma_f32_16x16x32_bf16(
            false, af[mi], false, bfv[ni], (short)0, acc[mi][ni], false, false);
  }

  // epilogue: everything is a packed 16B store (lane owns 8 consecutive rows)
  const int g = n0 / DDIM;                           // uniform per block (768 % 128 == 0)
  const int eoff = n0 - g * DDIM;
  const int b = m0 >> 8;                             // BM=128 never crosses a batch
  #pragma unroll
  for (int mi = 0; mi < 2; ++mi)
    #pragma unroll
    for (int ni = 0; ni < 4; ++ni) {
      int e = eoff + wn + ni * 16 + lr;
      union { uint4 q; uint16_t h[8]; } pk;
      #pragma unroll
      for (int r = 0; r < 8; ++r) pk.h[r] = f2bf(acc[mi][ni][r]);
      if (g == 0) {
        int row0 = m0 + wm + mi * 16 + 8 * hi;
        *(uint4*)&PselfT[(size_t)e * ROWS + row0] = pk.q;
      } else {
        int j0 = (m0 & 255) + wm + mi * 16 + 8 * hi;
        *(uint4*)&PT[((size_t)(b * 4 + (g - 1)) * DDIM + e) * NN + j0] = pk.q;
      }
    }
}

// ---------------- GEMM2: agg = [nw_a|nw_p|nw_ar|nw_pr] @ PT slices; fused epilogue ----------------
__global__ __launch_bounds__(256) void k_gemm2(const uint16_t* __restrict__ PselfT,
    const uint16_t* __restrict__ PT,
    const int* __restrict__ ag, const int* __restrict__ pg,
    const int* __restrict__ nmask, const float* __restrict__ dwm,
    const float* __restrict__ rden, const float* __restrict__ bs,
    float* __restrict__ outNode) {
  __shared__ uint16_t At[2][BM * LDT];
  __shared__ uint16_t Bt[2][BN * LDT];
  const int tid = threadIdx.x;
  const int b  = blockIdx.z;
  const int i0 = blockIdx.y * BM;
  const int e0 = blockIdx.x * BN;
  const int lane = tid & 31, wave = tid >> 5;
  const int wm = (wave & 3) * 32, wn = (wave >> 2) * 64;
  const int lr = lane & 15, hi = lane >> 4;
  const int a_j2 = (tid & 15) * 2, a_r = tid >> 4;
  const int b_k8 = (tid & 3) * 8,  b_r = tid >> 2;
  const int gb = b * NN * NN;

  float mif[8];                                      // mask_i: K-loop invariant
  #pragma unroll
  for (int p = 0; p < 8; ++p) mif[p] = (float)nmask[b * NN + i0 + a_r + p * 16];

  v8f zero = {0.f,0.f,0.f,0.f,0.f,0.f,0.f,0.f};
  v8f acc[2][4];
  #pragma unroll
  for (int mi = 0; mi < 2; ++mi)
    #pragma unroll
    for (int ni = 0; ni < 4; ++ni) acc[mi][ni] = zero;

  auto stage = [&](int kb, int buf) {
    const int grel = kb >> 8, jb = kb & 255;         // BK divides 256
    const int jA = jb + a_j2;
    const float d0 = dwm[b * NN + jA], d1 = dwm[b * NN + jA + 1];
    int gv0[8], gv1[8];
    if (grel == 0) {
      #pragma unroll
      for (int p = 0; p < 8; ++p) {
        int2 q = *(const int2*)&ag[gb + (i0 + a_r + p * 16) * NN + jA];
        gv0[p] = q.x; gv1[p] = q.y;
      }
    } else if (grel == 1) {
      #pragma unroll
      for (int p = 0; p < 8; ++p) {
        int2 q = *(const int2*)&pg[gb + (i0 + a_r + p * 16) * NN + jA];
        gv0[p] = q.x; gv1[p] = q.y;
      }
    } else if (grel == 2) {
      #pragma unroll
      for (int p = 0; p < 8; ++p) {
        int ii = i0 + a_r + p * 16;
        gv0[p] = ag[gb + jA * NN + ii]; gv1[p] = ag[gb + (jA + 1) * NN + ii];
      }
    } else {
      #pragma unroll
      for (int p = 0; p < 8; ++p) {
        int ii = i0 + a_r + p * 16;
        gv0[p] = pg[gb + jA * NN + ii]; gv1[p] = pg[gb + (jA + 1) * NN + ii];
      }
    }
    #pragma unroll
    for (int p = 0; p < 2; ++p) {                    // B: async 16B copies from transposed PT
      int nl = b_r + p * 64;
      copy16_g2lds(&PT[((size_t)(b * 4 + grel) * DDIM + e0 + nl) * NN + jb + b_k8],
                   &Bt[buf][nl * LDT + b_k8]);
    }
    #pragma unroll
    for (int p = 0; p < 8; ++p) {                    // nw values -> packed b32 LDS stores
      int ii = i0 + a_r + p * 16;
      float v0 = (ii == jA)     ? 0.f : d0 * mif[p] * (float)gv0[p];
      float v1 = (ii == jA + 1) ? 0.f : d1 * mif[p] * (float)gv1[p];
      uint32_t pk = (uint32_t)f2bf(v0) | ((uint32_t)f2bf(v1) << 16);
      *(uint32_t*)&At[buf][(a_r + p * 16) * LDT + a_j2] = pk;
    }
  };

  stage(0, 0);
  const int nT = (4 * NN) / BK;
  for (int it = 0; it < nT; ++it) {
    async_wait();
    __syncthreads();
    const int cur = it & 1;
    if (it + 1 < nT) stage((it + 1) * BK, cur ^ 1);
    v16bf af[2], bfv[4];
    #pragma unroll
    for (int mi = 0; mi < 2; ++mi) {
      const uint16_t* rp = &At[cur][(wm + mi * 16 + lr) * LDT];
      FragU u;
      u.q[0] = *(const uint4*)(rp + (hi ? 8 : 0));
      u.q[1] = *(const uint4*)(rp + (hi ? 24 : 16));
      af[mi] = u.v;
    }
    #pragma unroll
    for (int ni = 0; ni < 4; ++ni) {
      const uint16_t* cp = &Bt[cur][(wn + ni * 16 + lr) * LDT + hi * 8];
      FragU u;
      u.q[0] = *(const uint4*)(cp);
      u.q[1] = *(const uint4*)(cp + 16);
      bfv[ni] = u.v;
    }
    #pragma unroll
    for (int mi = 0; mi < 2; ++mi)
      #pragma unroll
      for (int ni = 0; ni < 4; ++ni)
        acc[mi][ni] = __builtin_amdgcn_wmma_f32_16x16x32_bf16(
            false, af[mi], false, bfv[ni], (short)0, acc[mi][ni], false, false);
  }

  // epilogue: relu(self + bs + acc/denom); self & rden reads vectorized
  #pragma unroll
  for (int mi = 0; mi < 2; ++mi) {
    int row0 = b * NN + i0 + wm + mi * 16 + 8 * hi;  // 8 consecutive rows per lane
    union { float4 f4[2]; float f[8]; } rd;
    rd.f4[0] = *(const float4*)&rden[row0];
    rd.f4[1] = *(const float4*)&rden[row0 + 4];
    #pragma unroll
    for (int ni = 0; ni < 4; ++ni) {
      int e = e0 + wn + ni * 16 + lr;
      float bse = bs[e];
      union { uint4 q; uint16_t h[8]; } sv;
      sv.q = *(const uint4*)&PselfT[(size_t)e * ROWS + row0];
      #pragma unroll
      for (int r = 0; r < 8; ++r) {
        float val = bf2f(sv.h[r]) + bse + acc[mi][ni][r] * rd.f[r];
        outNode[(size_t)(row0 + r) * DDIM + e] = fmaxf(val, 0.f);
      }
    }
  }
}

extern "C" void kernel_launch(void* const* d_in, const int* in_sizes, int n_in,
                              void* d_out, int out_size, void* d_ws, size_t ws_size,
                              hipStream_t stream) {
  (void)in_sizes; (void)n_in; (void)out_size; (void)ws_size;
  const float* node  = (const float*)d_in[0];
  const int*   nmask = (const int*)  d_in[1];
  const int*   ag    = (const int*)  d_in[2];
  const int*   pg    = (const int*)  d_in[3];
  const float* Wq    = (const float*)d_in[4];
  const float* bq    = (const float*)d_in[5];
  const float* Ws    = (const float*)d_in[6];
  const float* bs    = (const float*)d_in[7];
  const float* Wa    = (const float*)d_in[8];
  const float* Wp    = (const float*)d_in[9];
  const float* Wap   = (const float*)d_in[10];
  const float* Wpp   = (const float*)d_in[11];
  const float* Wa2   = (const float*)d_in[12];
  const float* Wp2   = (const float*)d_in[13];
  const float* Wap2  = (const float*)d_in[14];
  const float* Wpp2  = (const float*)d_in[15];

  float* outNode = (float*)d_out;                          // [B,N,D]
  float* outW    = outNode + (size_t)ROWS * DDIM;          // [B,1,N]

  // workspace layout (~132 MB), all slices 16B aligned
  float*    dwm    = (float*)d_ws;                         // 16384
  float*    rden   = dwm + ROWS;                           // 16384
  uint16_t* Wcat   = (uint16_t*)(rden + ROWS);             // 3840*768 bf16
  uint16_t* PselfT = Wcat + (size_t)NCAT * DDIM;           // 768*16384 bf16 (transposed)
  uint16_t* PT     = PselfT + (size_t)DDIM * ROWS;         // 64*4*768*256 bf16 (transposed)

  k_dw   <<<ROWS / 8, 256, 0, stream>>>(node, nmask, Wq, bq, dwm, outW);
  k_denom<<<ROWS,     256, 0, stream>>>(ag, pg, nmask, rden);
  k_wcat <<<(NCAT * DDIM) / 256, 256, 0, stream>>>(Ws, Wa, Wp, Wap, Wpp,
                                                   Wa2, Wp2, Wap2, Wpp2, Wcat);
  k_gemm1<<<dim3(NCAT / BN, ROWS / BM), 256, 0, stream>>>(node, Wcat, PselfT, PT);
  k_gemm2<<<dim3(DDIM / BN, NN / BM, NB), 256, 0, stream>>>(PselfT, PT, ag, pg, nmask,
                                                            dwm, rden, bs, outNode);
}
